// TranslateLayer_82446192214723
// MI455X (gfx1250) — compile-verified, matrix operand order checked by reading, not run on hardware
//
#include <hip/hip_runtime.h>
#include <hip/hip_bf16.h>

// ---------------------------------------------------------------------------
// CDNA5 (gfx1250) implementation.
//   conv1/conv2 of the condense net -> implicit GEMM on v_wmma_f32_16x16x32_f16
//   (f16 inputs, f32 accumulate); dynamic 9-tap filter -> bandwidth-bound
//   float4 pass. All heavy loads 16B aligned. Halo handling is branchless:
//   out-of-bounds A rows read from a zeroed scratch row so the K-loop is pure
//   clause'd global_load_b128 + v_wmma (no exec-mask branches).
// ---------------------------------------------------------------------------

typedef __attribute__((ext_vector_type(16))) _Float16 v16h;
typedef __attribute__((ext_vector_type(8)))  _Float16 v8h;
typedef __attribute__((ext_vector_type(8)))  float    v8f;

union F16x16 { v16h v; v8h h[2]; };

#define Bn   8
#define NH   128
#define NW   128
#define CC   128   // ctrl channels
#define C1P  64    // conv1 out channels padded (50 -> 64)
#define C2P  32    // conv2 out channels padded (18 -> 32)
#define HH   512
#define WW   512

// ---------------------------------------------------------------------------
// Kernel 0: zero the 128-half scratch row used for out-of-bounds halo reads.
// ---------------------------------------------------------------------------
__global__ void zfill_kernel(_Float16* __restrict__ zrow) {
    zrow[threadIdx.x] = (_Float16)0.0f;
}

// ---------------------------------------------------------------------------
// Kernel 1: transpose ctrl NCHW f32 -> NHWC f16  (LDS-tiled, coalesced)
// src [b][ci][y][x], dst [b][y][x][ci]; all of ci,y,x = 128.
// grid (x_tiles=4, ci_tiles=4, b*y=1024), block (32,8)
// ---------------------------------------------------------------------------
__global__ void pack_ctrl_kernel(const float* __restrict__ src,
                                 _Float16* __restrict__ dst) {
    __shared__ float tile[32][33];
    const int by = blockIdx.z;
    const int b  = by >> 7;
    const int y  = by & 127;
    const int x0 = blockIdx.x * 32;
    const int c0 = blockIdx.y * 32;
#pragma unroll
    for (int i = 0; i < 32; i += 8) {
        const int ci = c0 + threadIdx.y + i;
        tile[threadIdx.y + i][threadIdx.x] =
            src[((b * CC + ci) * NH + y) * NW + x0 + threadIdx.x];
    }
    __syncthreads();
#pragma unroll
    for (int i = 0; i < 32; i += 8) {
        const int x  = x0 + threadIdx.y + i;
        const int ci = c0 + threadIdx.x;
        dst[((b * NH + y) * NW + x) * CC + ci] =
            (_Float16)tile[threadIdx.x][threadIdx.y + i];
    }
}

// ---------------------------------------------------------------------------
// Kernel 2: pack w1 (50,128,3,3) OIHW f32 -> [tap=9][co=64][ci=128] f16
// (co padded with zeros; stored "column-major" so B-fragment loads mirror A)
// ---------------------------------------------------------------------------
__global__ void pack_w1_kernel(const float* __restrict__ w1,
                               _Float16* __restrict__ w1p) {
    const int id = blockIdx.x * 256 + threadIdx.x;
    if (id >= 9 * C1P * CC) return;
    const int ci  = id % CC;
    const int co  = (id / CC) % C1P;
    const int tap = id / (CC * C1P);
    float v = 0.f;
    if (co < 50) v = w1[(co * CC + ci) * 9 + tap];
    w1p[id] = (_Float16)v;
}

// ---------------------------------------------------------------------------
// Kernel 3: pack w2 (18,50,3,3) f32 -> [tap=9][co=32][ci=64] f16 (zero padded)
// ---------------------------------------------------------------------------
__global__ void pack_w2_kernel(const float* __restrict__ w2,
                               _Float16* __restrict__ w2p) {
    const int id = blockIdx.x * 256 + threadIdx.x;
    if (id >= 9 * C2P * C1P) return;
    const int ci  = id % C1P;
    const int co  = (id / C1P) % C2P;
    const int tap = id / (C1P * C2P);
    float v = 0.f;
    if (co < 18 && ci < 50) v = w2[(co * 50 + ci) * 9 + tap];
    w2p[id] = (_Float16)v;
}

// ---------------------------------------------------------------------------
// A/B fragment load helper: row-major [row][K] with 256B-aligned rows.
// Lane layout per ISA 7.12.2 (16-bit A 16x32): lane holds row = lane&15,
// halves K = k0..k0+7 (VGPR0-3) and k0+16..k0+23 (VGPR4-7), k0 = 8*(lane>=16).
// ---------------------------------------------------------------------------
__device__ __forceinline__ F16x16 load_frag(const _Float16* __restrict__ p) {
    F16x16 f;
    f.h[0] = *(const v8h*)(p);
    f.h[1] = *(const v8h*)(p + 16);
    return f;
}

// ---------------------------------------------------------------------------
// Kernel 4: conv1 (128->50, 3x3, pad 1) + bias + ReLU + BN, implicit GEMM.
// Block = 128 threads (4 waves). M-tile = 16 pixels of one row; wave w owns
// co tile [16w, 16w+16). K loop: 9 taps x 4 chunks of 32 -> 36 WMMAs/wave.
// Out-of-bounds halo rows read from zrow (branchless). Output: h NHWC f16
// [b][y][x][64], co>=50 zeroed.
// ---------------------------------------------------------------------------
__global__ void conv1_kernel(const _Float16* __restrict__ ctrlp,
                             const _Float16* __restrict__ w1p,
                             const _Float16* __restrict__ zrow,
                             const float* __restrict__ b1,
                             const float* __restrict__ gamma,
                             const float* __restrict__ beta,
                             const float* __restrict__ mean,
                             const float* __restrict__ var,
                             _Float16* __restrict__ hbuf) {
    const int lane = threadIdx.x & 31;
    const int wave = threadIdx.x >> 5;          // co n-tile
    const int m    = lane & 15;
    const int hi   = lane >> 4;
    const int k0   = hi * 8;

    const int bid = blockIdx.x;                  // 8 * 128 * 8 = 8192 blocks
    const int xt  = bid & 7;
    const int y   = (bid >> 3) & 127;
    const int b   = bid >> 10;
    const int x0  = xt * 16;

    const int arow_co = wave * 16 + m;           // B "row" index (= co)
    v8f acc = {};
    const _Float16* zbase = zrow + k0;

#pragma unroll
    for (int t = 0; t < 9; ++t) {
        const int dy = t / 3 - 1, dx = t % 3 - 1;
        const int yy = y + dy;
        const int xx = x0 + m + dx;
        const bool aval = ((unsigned)yy < (unsigned)NH) && ((unsigned)xx < (unsigned)NW);
        const _Float16* arow = ctrlp + (((size_t)(b * NH + yy) * NW + xx) * CC) + k0;
        const _Float16* abase = aval ? arow : zbase;   // branchless halo
        const _Float16* bbase = w1p + ((size_t)(t * C1P + arow_co) * CC) + k0;
#pragma unroll
        for (int c = 0; c < 4; ++c) {
            F16x16 a  = load_frag(abase + c * 32);
            F16x16 bf = load_frag(bbase + c * 32);
            acc = __builtin_amdgcn_wmma_f32_16x16x32_f16(
                false, a.v, false, bf.v, (short)0, acc, false, false);
        }
    }

    // Epilogue: C layout -> n = lane&15, M = r + 8*hi
    const int n  = lane & 15;
    const int co = wave * 16 + n;
    const bool live = co < 50;
    const float bias = live ? b1[co] : 0.f;
    const float sc   = live ? gamma[co] * rsqrtf(var[co] + 1e-5f) : 0.f;
    const float mn   = live ? mean[co] : 0.f;
    const float bt   = live ? beta[co] : 0.f;
#pragma unroll
    for (int r = 0; r < 8; ++r) {
        const int mm = r + hi * 8;
        float v = acc[r] + bias;
        v = fmaxf(v, 0.f);
        v = (v - mn) * sc + bt;
        if (!live) v = 0.f;
        hbuf[(((size_t)(b * NH + y) * NW + x0 + mm) * C1P) + co] = (_Float16)v;
    }
}

// ---------------------------------------------------------------------------
// Kernel 5: conv2 (50->18 padded 64->32, 3x3, pad 1) + b2, implicit GEMM.
// Block = 64 threads (2 waves). 9 taps x 2 chunks -> 18 WMMAs/wave.
// Output: cond NHWC f32 [b][y][x][18].
// ---------------------------------------------------------------------------
__global__ void conv2_kernel(const _Float16* __restrict__ hbuf,
                             const _Float16* __restrict__ w2p,
                             const _Float16* __restrict__ zrow,
                             const float* __restrict__ b2,
                             float* __restrict__ cond) {
    const int lane = threadIdx.x & 31;
    const int wave = threadIdx.x >> 5;
    const int m    = lane & 15;
    const int hi   = lane >> 4;
    const int k0   = hi * 8;

    const int bid = blockIdx.x;
    const int xt  = bid & 7;
    const int y   = (bid >> 3) & 127;
    const int b   = bid >> 10;
    const int x0  = xt * 16;

    const int arow_co = wave * 16 + m;
    v8f acc = {};
    const _Float16* zbase = zrow + k0;

#pragma unroll
    for (int t = 0; t < 9; ++t) {
        const int dy = t / 3 - 1, dx = t % 3 - 1;
        const int yy = y + dy;
        const int xx = x0 + m + dx;
        const bool aval = ((unsigned)yy < (unsigned)NH) && ((unsigned)xx < (unsigned)NW);
        const _Float16* arow = hbuf + (((size_t)(b * NH + yy) * NW + xx) * C1P) + k0;
        const _Float16* abase = aval ? arow : zbase;   // branchless halo
        const _Float16* bbase = w2p + ((size_t)(t * C2P + arow_co) * C1P) + k0;
#pragma unroll
        for (int c = 0; c < 2; ++c) {
            F16x16 a  = load_frag(abase + c * 32);
            F16x16 bf = load_frag(bbase + c * 32);
            acc = __builtin_amdgcn_wmma_f32_16x16x32_f16(
                false, a.v, false, bf.v, (short)0, acc, false, false);
        }
    }

    const int n  = lane & 15;
    const int co = wave * 16 + n;
    if (co < 18) {
        const float bias = b2[co];
#pragma unroll
        for (int r = 0; r < 8; ++r) {
            const int mm = r + hi * 8;
            cond[(((size_t)(b * NH + y) * NW + x0 + mm) * 18) + co] = acc[r] + bias;
        }
    }
}

// ---------------------------------------------------------------------------
// Kernel 6: dynamic per-cell 9-tap filter (bandwidth bound).
// Thread handles one (b, ch, y, cellx) -> 4 contiguous outputs (float4).
// Tap shifts are multiples of 4 floats -> all image loads 16B aligned.
// ---------------------------------------------------------------------------
__global__ void apply_kernel(const float* __restrict__ img,
                             const float* __restrict__ cond,
                             float* __restrict__ out) {
    const int tid = blockIdx.x * 256 + threadIdx.x;   // 8*6*512*128 threads
    const int cx   = tid & 127;
    const int y    = (tid >> 7) & 511;
    const int rest = tid >> 16;                        // b*6 + ch
    const int ch   = rest % 6;
    const int b    = rest / 6;

    const int g  = ch / 3;
    const int cy = y >> 2;
    const float* cp = cond + (((size_t)(b * NH + cy) * NW + cx) * 18) + g * 9;

    float w[9];
#pragma unroll
    for (int k = 0; k < 9; ++k) w[k] = cp[k];

    const float* ibase = img + ((size_t)(b * 6 + ch) * HH) * WW;
    float o0 = 0.f, o1 = 0.f, o2 = 0.f, o3 = 0.f;
#pragma unroll
    for (int i = 0; i < 3; ++i) {
        const int xx = cx * 4 + (i - 1) * 4;
        if (xx < 0 || xx >= WW) continue;
#pragma unroll
        for (int j = 0; j < 3; ++j) {
            const int yy = y + (j - 1) * 4;
            if (yy < 0 || yy >= HH) continue;
            const float4 v = *(const float4*)(ibase + (size_t)yy * WW + xx);
            const float wk = w[i * 3 + j];
            o0 += v.x * wk; o1 += v.y * wk; o2 += v.z * wk; o3 += v.w * wk;
        }
    }
    float4 r; r.x = o0; r.y = o1; r.z = o2; r.w = o3;
    *(float4*)(out + ((size_t)(b * 6 + ch) * HH + y) * WW + cx * 4) = r;
}

// ---------------------------------------------------------------------------
extern "C" void kernel_launch(void* const* d_in, const int* in_sizes, int n_in,
                              void* d_out, int out_size, void* d_ws, size_t ws_size,
                              hipStream_t stream) {
    (void)in_sizes; (void)n_in; (void)out_size; (void)ws_size;

    const float* image = (const float*)d_in[0];
    const float* ctrl  = (const float*)d_in[1];
    const float* w1    = (const float*)d_in[2];
    const float* b1    = (const float*)d_in[3];
    const float* gamma = (const float*)d_in[4];
    const float* beta  = (const float*)d_in[5];
    const float* mean  = (const float*)d_in[6];
    const float* var   = (const float*)d_in[7];
    const float* w2    = (const float*)d_in[8];
    const float* b2    = (const float*)d_in[9];

    char* ws = (char*)d_ws;
    size_t off = 0;
    _Float16* zrow  = (_Float16*)(ws + off); off += 256;                             // 128 halves
    _Float16* ctrlp = (_Float16*)(ws + off); off += (size_t)Bn * NH * NW * CC * 2;   // 33.6 MB
    _Float16* w1p   = (_Float16*)(ws + off); off += (size_t)9 * C1P * CC * 2;        // 147 KB
    _Float16* w2p   = (_Float16*)(ws + off); off += (size_t)9 * C2P * C1P * 2;       // 37 KB
    _Float16* hbuf  = (_Float16*)(ws + off); off += (size_t)Bn * NH * NW * C1P * 2;  // 16.8 MB
    float*    cond  = (float*)(ws + off);    off += (size_t)Bn * NH * NW * 18 * 4;   // 9.4 MB

    zfill_kernel<<<1, 128, 0, stream>>>(zrow);
    pack_ctrl_kernel<<<dim3(4, 4, Bn * NH), dim3(32, 8), 0, stream>>>(ctrl, ctrlp);
    pack_w1_kernel<<<(9 * C1P * CC + 255) / 256, 256, 0, stream>>>(w1, w1p);
    pack_w2_kernel<<<(9 * C2P * C1P + 255) / 256, 256, 0, stream>>>(w2, w2p);

    conv1_kernel<<<Bn * NH * (NW / 16), 128, 0, stream>>>(
        ctrlp, w1p, zrow, b1, gamma, beta, mean, var, hbuf);
    conv2_kernel<<<Bn * NH * (NW / 16), 64, 0, stream>>>(hbuf, w2p, zrow, b2, cond);

    const int total = Bn * 6 * HH * (WW / 4);
    apply_kernel<<<total / 256, 256, 0, stream>>>(image, cond, (float*)d_out);
}
